// FullAttention_86955907874884
// MI455X (gfx1250) — matrix operand order repaired and checked by reading, not compiled, not run on hardware
//
#include <hip/hip_runtime.h>

typedef __attribute__((ext_vector_type(16))) _Float16 v16h;
typedef __attribute__((ext_vector_type(8)))  _Float16 v8h;
typedef __attribute__((ext_vector_type(2)))  __fp16   v2hf;   // cvt_pkrtz return type
typedef __attribute__((ext_vector_type(8)))  float    v8f;
typedef __attribute__((ext_vector_type(4)))  unsigned int u32x4;
typedef __attribute__((ext_vector_type(8)))  int          i32x8;
typedef __attribute__((ext_vector_type(4)))  int          i32x4;

#define B_  4
#define L_  2048
#define H_  8
#define E_  64
#define QT  64        // query rows per block (4 waves x 16 rows)
#define KT  32        // key columns per inner iteration
#define KSTRIDE 68    // f32 row stride of TDM-staged K tile (64 + 4 pad DWORDs)
#define VPAD 40       // padded row stride (halves) for V^T tile in LDS
#define PPAD 40       // padded row stride (halves) for P scratch in LDS

__device__ __forceinline__ v16h cat16(v8h lo, v8h hi) {
  return __builtin_shufflevector(lo, hi, 0,1,2,3,4,5,6,7,8,9,10,11,12,13,14,15);
}

// 16 x f32 -> v16h using v_cvt_pkrtz_f16_f32 (convert+pack, 8 VALU ops)
__device__ __forceinline__ v16h cvt16(const float* f) {
  v16h r;
#pragma unroll
  for (int i = 0; i < 8; ++i) {
    v2hf p = __builtin_amdgcn_cvt_pkrtz(f[2*i], f[2*i+1]);
    r[2*i]   = (_Float16)p[0];
    r[2*i+1] = (_Float16)p[1];
  }
  return r;
}

// 16-lane row reduction step via DPP16 (pure VALU, no LDS pipeline)
#define DPP_MAX(v, ctrl)                                                     \
  v = fmaxf(v, __int_as_float(__builtin_amdgcn_update_dpp(                   \
        __float_as_int(v), __float_as_int(v), (ctrl), 0xf, 0xf, true)))

// Raw hardware exp2: args are always <= 0 here; huge-negative flushes to 0,
// which is the masking semantics we want (no ldexp fixup sequences).
#define EXP2(x) __builtin_amdgcn_exp2f(x)

// Issue a TDM DMA: 32 rows x 64 f32, row stride 512 f32, global -> LDS,
// inserting 4 pad DWORDs after every 64 DWORDs (row stride 68 DW in LDS).
__device__ __forceinline__ void tdm_load_k(const float* gptr, unsigned lds_off) {
  unsigned long long ga = (unsigned long long)gptr;
  u32x4 g0 = { 1u,                                   // count=1 (valid D#)
               lds_off,                              // lds_addr
               (unsigned)(ga & 0xffffffffu),         // global_addr[31:0]
               (unsigned)((ga >> 32) & 0x01ffffffu)  // global_addr[56:32]
                 | 0x80000000u };                    // type=2 ("image")
  i32x8 g1 = { 0x07520000,   // data_size=4B, pad_enable, interval=64DW, amount=4DW
               0x00400000,   // tensor_dim0 = 64 (low16 @ bits 63:48)
               0x00200000,   // tensor_dim1 = 32 (low16 @ bits 95:80)
               0x00400000,   // tile_dim0  = 64 (bits 127:112)
               32,           // tile_dim1  = 32 (bits 143:128)
               512,          // tensor_dim0_stride = 512 elements
               0, 0 };
  i32x4 z4 = {0,0,0,0};
#if __has_include(<hip/amd_detail/amd_gfx1250_TDM.h>)
  i32x8 z8 = {0,0,0,0,0,0,0,0};
  __builtin_amdgcn_tensor_load_to_lds(g0, g1, z4, z4, z8, 0);
#else
  __builtin_amdgcn_tensor_load_to_lds(g0, g1, z4, z4, 0);
#endif
}

__launch_bounds__(128)
__global__ void fa_causal_kernel(const float* __restrict__ Qg,
                                 const float* __restrict__ Kg,
                                 const float* __restrict__ Vg,
                                 float* __restrict__ Og)
{
  __shared__ __align__(16) float    sK32[2][KT * KSTRIDE]; // double-buffered TDM K
  __shared__ __align__(16) _Float16 sVT[E_ * VPAD];        // [d][key] f16 (transposed)
  __shared__ __align__(16) _Float16 sP[4][16 * PPAD];      // per-wave P scratch

  const int tid  = threadIdx.x;
  const int lane = tid & 31;
  const int hl   = lane >> 4;   // lane half (0/1)
  const int l16  = lane & 15;
  // SGPR-provable "am I wave 0": guarantees a scalar branch around the TDM op
  // (TDM ignores EXEC, so we must actually branch, not just mask).
  const bool is_w0 = (__builtin_amdgcn_readfirstlane((int)(tid >> 5)) == 0);
  const int wv = tid >> 5;

  const int qtiles = L_ / QT;
  const int qt = blockIdx.x % qtiles;
  const int bh = blockIdx.x / qtiles;
  const int h  = bh % H_;
  const int b  = bh / H_;
  const int q0 = qt * QT;

  const size_t sL = (size_t)H_ * E_;   // stride between sequence positions
  const float* Qb = Qg + ((size_t)b * L_) * sL + (size_t)h * E_;
  const float* Kb = Kg + ((size_t)b * L_) * sL + (size_t)h * E_;
  const float* Vb = Vg + ((size_t)b * L_) * sL + (size_t)h * E_;
  float*       Ob = Og + ((size_t)b * L_) * sL + (size_t)h * E_;

  const int myrow0 = q0 + wv * 16;
  const float C = 0.18033688011112042f;  // (1/sqrt(64)) * log2(e), folded into Q

  // ---- Q in WMMA A layout, pre-scaled by C (f32 -> f16), kept in registers ----
  v16h aQ[2];
  {
    const float* qr = Qb + (size_t)(myrow0 + l16) * sL;
    const int eo = hl * 8;
#pragma unroll
    for (int es = 0; es < 2; ++es) {
      float4 x0 = *(const float4*)(qr + es*32 + eo);
      float4 x1 = *(const float4*)(qr + es*32 + eo + 4);
      float4 x2 = *(const float4*)(qr + es*32 + 16 + eo);
      float4 x3 = *(const float4*)(qr + es*32 + 16 + eo + 4);
      float f[16] = { x0.x*C,x0.y*C,x0.z*C,x0.w*C, x1.x*C,x1.y*C,x1.z*C,x1.w*C,
                      x2.x*C,x2.y*C,x2.z*C,x2.w*C, x3.x*C,x3.y*C,x3.z*C,x3.w*C };
      aQ[es] = cvt16(f);
    }
  }

  // Ones B-fragment: row-sums of P via one extra WMMA into acc[4]
  v16h ones;
#pragma unroll
  for (int i = 0; i < 16; ++i) ones[i] = (_Float16)1.0f;

  // acc[0..3] = O d-tiles; acc[4] = softmax denominator (row sums)
  v8f acc[5];
#pragma unroll
  for (int a5 = 0; a5 < 5; ++a5)
    acc[a5] = v8f{0.f,0.f,0.f,0.f,0.f,0.f,0.f,0.f};
  float mrow[8];
#pragma unroll
  for (int r = 0; r < 8; ++r) mrow[r] = -1e30f;

  const int rowmax = myrow0 + 15;
  const int ntiles = (q0 + QT) / KT;     // causal: keys up to q0+63 only
  const unsigned sK_lds0 = (unsigned)(unsigned long long)(void*)&sK32[0][0];
  const unsigned sK_lds1 = (unsigned)(unsigned long long)(void*)&sK32[1][0];

  // Prologue: start DMA of K tile 0 into buffer 0
  if (is_w0) tdm_load_k(Kb, sK_lds0);

  for (int kt = 0; kt < ntiles; ++kt) {
    const int j0  = kt * KT;
    const int buf = kt & 1;
    __syncthreads();                      // prior-iteration LDS reads complete
    // K tile kt was issued one iteration ago; its DMA ran under that compute.
    if (is_w0) __builtin_amdgcn_s_wait_tensorcnt(0);
    // ---- stage V tile transposed (d-major) f32 -> f16, all threads ----
    {
      const int j  = tid & 31;
      const int d0 = (tid >> 5) * 16;
      const float* vp = Vb + (size_t)(j0 + j) * sL + d0;
      float4 x0 = *(const float4*)(vp + 0);
      float4 x1 = *(const float4*)(vp + 4);
      float4 x2 = *(const float4*)(vp + 8);
      float4 x3 = *(const float4*)(vp + 12);
      float f[16] = { x0.x,x0.y,x0.z,x0.w, x1.x,x1.y,x1.z,x1.w,
                      x2.x,x2.y,x2.z,x2.w, x3.x,x3.y,x3.z,x3.w };
#pragma unroll
      for (int k = 0; k < 16; ++k) sVT[(d0 + k) * VPAD + j] = (_Float16)f[k];
      if (j0 + KT < L_) __builtin_prefetch(vp + (size_t)KT * sL, 0, 0);
    }
    __syncthreads();                      // publish K buf[kt&1] (TDM) + V^T
    // Kick off next tile's DMA into the other buffer: overlaps with compute.
    if (is_w0 && (kt + 1) < ntiles)
      tdm_load_k(Kb + (size_t)(j0 + KT) * sL, buf ? sK_lds0 : sK_lds1);

    if (j0 <= rowmax) {  // wave-uniform: skip fully-masked tiles, keep barriers
      // ---- scores: S = (C*Q) K^T ; 2 n-tiles of 16 keys, K-dim 64 in 2 steps ----
      v8f s[2];
#pragma unroll
      for (int nt = 0; nt < 2; ++nt) {
        s[nt] = v8f{0.f,0.f,0.f,0.f,0.f,0.f,0.f,0.f};
#pragma unroll
        for (int es = 0; es < 2; ++es) {
          // B 32x16: lane = key column, halves = contraction e (f32->f16 pkrtz)
          const float* kb = &sK32[buf][(nt*16 + l16) * KSTRIDE + es*32 + hl*8];
          float4 y0 = *(const float4*)(kb + 0);
          float4 y1 = *(const float4*)(kb + 4);
          float4 y2 = *(const float4*)(kb + 16);
          float4 y3 = *(const float4*)(kb + 20);
          float f[16] = { y0.x,y0.y,y0.z,y0.w, y1.x,y1.y,y1.z,y1.w,
                          y2.x,y2.y,y2.z,y2.w, y3.x,y3.y,y3.z,y3.w };
          v16h bK = cvt16(f);
          s[nt] = __builtin_amdgcn_wmma_f32_16x16x32_f16(
              false, aQ[es], false, bK, (short)0, s[nt], false, false);
        }
      }
      // ---- causal mask only on diagonal-intersecting tiles (wave-uniform) ----
      float t0[8], t1[8];
      if (j0 + KT - 1 > myrow0) {
#pragma unroll
        for (int r = 0; r < 8; ++r) {
          const int qr = myrow0 + r + hl*8;
          t0[r] = (j0 + l16      <= qr) ? s[0][r] : -1e30f;
          t1[r] = (j0 + 16 + l16 <= qr) ? s[1][r] : -1e30f;
        }
      } else {
#pragma unroll
        for (int r = 0; r < 8; ++r) { t0[r] = s[0][r]; t1[r] = s[1][r]; }
      }
      // ---- online softmax: DPP16 row-max, WMMA row-sum, raw v_exp_f32 ----
      float p0[8], p1[8];
#pragma unroll
      for (int r = 0; r < 8; ++r) {
        float mt = fmaxf(t0[r], t1[r]);
        DPP_MAX(mt, 0xB1);   // quad_perm(1,0,3,2)  : xor 1
        DPP_MAX(mt, 0x4E);   // quad_perm(2,3,0,1)  : xor 2
        DPP_MAX(mt, 0x141);  // ROW_HALF_MIRROR     : xor 4
        DPP_MAX(mt, 0x140);  // ROW_MIRROR          : xor 15 (== xor 8 here)
        const float mn = fmaxf(mrow[r], mt);
        const float al = EXP2(mrow[r] - mn);
        mrow[r] = mn;
        p0[r] = EXP2(t0[r] - mn);
        p1[r] = EXP2(t1[r] - mn);
#pragma unroll
        for (int a5 = 0; a5 < 5; ++a5) acc[a5][r] *= al;
      }
      // ---- P (D layout) -> per-wave LDS -> reload in A layout ----
      _Float16* pw = &sP[wv][0];
#pragma unroll
      for (int r = 0; r < 8; ++r) {
        const int rr = r + hl*8;
        pw[rr * PPAD + l16]      = (_Float16)p0[r];
        pw[rr * PPAD + 16 + l16] = (_Float16)p1[r];
      }
      const _Float16* pr = &sP[wv][l16 * PPAD + hl*8];
      v16h aP = cat16(*(const v8h*)pr, *(const v8h*)(pr + 16));
      // ---- O += P * V (4 d-tiles) ; denominator += P * ones ----
#pragma unroll
      for (int dt = 0; dt < 4; ++dt) {
        const _Float16* vb = &sVT[(dt*16 + l16) * VPAD + hl*8];
        v16h bV = cat16(*(const v8h*)vb, *(const v8h*)(vb + 16));
        acc[dt] = __builtin_amdgcn_wmma_f32_16x16x32_f16(
            false, aP, false, bV, (short)0, acc[dt], false, false);
      }
      acc[4] = __builtin_amdgcn_wmma_f32_16x16x32_f16(
          false, aP, false, ones, (short)0, acc[4], false, false);
    }
  }

  // ---- epilogue: normalize by row sum, store fp32 ----
#pragma unroll
  for (int r = 0; r < 8; ++r) {
    const float inv = 1.0f / acc[4][r];
    float* op = Ob + (size_t)(myrow0 + r + hl*8) * sL + l16;
#pragma unroll
    for (int dt = 0; dt < 4; ++dt)
      op[dt * 16] = acc[dt][r] * inv;
  }
}

extern "C" void kernel_launch(void* const* d_in, const int* in_sizes, int n_in,
                              void* d_out, int out_size, void* d_ws, size_t ws_size,
                              hipStream_t stream) {
  (void)in_sizes; (void)n_in; (void)out_size; (void)d_ws; (void)ws_size;
  const float* Q = (const float*)d_in[0];
  const float* K = (const float*)d_in[1];
  const float* V = (const float*)d_in[2];
  float*       O = (float*)d_out;
  dim3 grid(B_ * H_ * (L_ / QT));
  dim3 block(128);
  fa_causal_kernel<<<grid, block, 0, stream>>>(Q, K, V, O);
}
